// DynamicGraphModule_15144054686343
// MI455X (gfx1250) — matrix operand order repaired and checked by reading, not compiled, not run on hardware
//
#include <hip/hip_runtime.h>
#include <hip/hip_bf16.h>

// ---- problem dims (fixed by the reference) ----
#define NROW 4096      // N spans
#define INDIM 1024
#define HID 512
#define DDIM 512
#define OUTD 512

typedef _Float16 v16h __attribute__((ext_vector_type(16)));
typedef _Float16 v8h  __attribute__((ext_vector_type(8)));
typedef float    v8f  __attribute__((ext_vector_type(8)));

// ------------------------------------------------------------------
// K0a: fp32 -> fp16 convert (grid-stride), used for X
// ------------------------------------------------------------------
__global__ void cvt_f32_f16(const float* __restrict__ src, _Float16* __restrict__ dst, int n) {
    for (int i = blockIdx.x * blockDim.x + threadIdx.x; i < n; i += gridDim.x * blockDim.x)
        dst[i] = (_Float16)src[i];
}

// ------------------------------------------------------------------
// K0b: convert + pack W_c1[1024x512] into WMMA-B fragment layout:
//   Bp[((kkblk*512 + n)*2 + half)*16 + i] = (f16) W[(kkblk*32 + half*16 + i)*512 + n]
// so a GEMM lane reads its whole 16-half B fragment as one contiguous 32B v16h.
// ------------------------------------------------------------------
__global__ void pack_W1(const float* __restrict__ W, _Float16* __restrict__ Bp) {
    int idx = blockIdx.x * blockDim.x + threadIdx.x;      // coalesced read (n fastest)
    if (idx >= INDIM * HID) return;
    int k = idx >> 9;                                     // row  (0..1023)
    int n = idx & 511;                                    // col  (0..511)
    int kkblk = k >> 5;
    int rem   = k & 31;
    int half  = rem >> 4;
    int i     = rem & 15;
    Bp[((((size_t)kkblk * HID + n) * 2 + half) << 4) + i] = (_Float16)W[idx];
}

// ------------------------------------------------------------------
// K1: H = relu(Xh[4096x1024] @ W1[1024x512] + b_c1) via v_wmma_f32_16x16x32_f16.
// Each wave32 computes a 16(M) x 64(N) strip: one A fragment feeds 4 WMMAs.
// B fragments come pre-packed (contiguous 32B per lane).
// ------------------------------------------------------------------
__global__ void gemm_relu_wmma(const _Float16* __restrict__ A,    // [M,K] row-major f16
                               const _Float16* __restrict__ Bp,   // packed B fragments
                               const float*    __restrict__ bias, // [N]
                               float*          __restrict__ H)    // [M,N] f32 out
{
    const int M = NROW, K = INDIM, N = HID;
    const int wid  = (blockIdx.x * blockDim.x + threadIdx.x) >> 5;
    const int lane = threadIdx.x & 31;
    if (wid >= (M / 16) * (N / 64)) return;
    const int m0 = (wid >> 3) * 16;            // 256 M-tiles
    const int n0 = (wid & 7) * 64;             // 8 N-strips of 64

    const int mrow = lane & 15;
    const int hi   = lane >> 4;                // lane half selects K sub-range

    v8f acc0 = {}, acc1 = {}, acc2 = {}, acc3 = {};
    const _Float16* arow = A + (size_t)(m0 + mrow) * K + hi * 16;
    // per-lane packed-B base for column n0 + (lane&15); subtile s is +s*512 halves
    const _Float16* bcol0 = Bp + ((((size_t)(n0 + (lane & 15))) * 2 + hi) << 4);

    #pragma unroll 2
    for (int kk = 0; kk < K; kk += 32) {
        // A fragment: lane<16 -> K {kk..kk+15}; lane>=16 -> K {kk+16..kk+31}
        const _Float16* ap = arow + kk;
        v8h alo = *(const v8h*)(ap);
        v8h ahi = *(const v8h*)(ap + 8);
        v16h a;
        #pragma unroll
        for (int i = 0; i < 8; ++i) { a[i] = alo[i]; a[8 + i] = ahi[i]; }
        if (kk + 64 < K) __builtin_prefetch(arow + kk + 64, 0, 1);

        const _Float16* bp = bcol0 + ((size_t)(kk >> 5) * HID << 5); // kkblk*512*2*16 halves
        v16h b0 = *(const v16h*)(bp);
        v16h b1 = *(const v16h*)(bp + 512);
        v16h b2 = *(const v16h*)(bp + 1024);
        v16h b3 = *(const v16h*)(bp + 1536);

        acc0 = __builtin_amdgcn_wmma_f32_16x16x32_f16(false, a, false, b0, (short)0, acc0, false, false);
        acc1 = __builtin_amdgcn_wmma_f32_16x16x32_f16(false, a, false, b1, (short)0, acc1, false, false);
        acc2 = __builtin_amdgcn_wmma_f32_16x16x32_f16(false, a, false, b2, (short)0, acc2, false, false);
        acc3 = __builtin_amdgcn_wmma_f32_16x16x32_f16(false, a, false, b3, (short)0, acc3, false, false);
    }

    // D layout: VGPR r -> row r (lanes 0-15) / row r+8 (lanes 16-31); col = lane&15
    const int rbase = m0 + hi * 8;
    v8f accs[4] = {acc0, acc1, acc2, acc3};
    #pragma unroll
    for (int s = 0; s < 4; ++s) {
        const int col  = n0 + s * 16 + (lane & 15);
        const float bv = bias[col];
        #pragma unroll
        for (int r = 0; r < 8; ++r) {
            float v = accs[s][r] + bv;
            H[(size_t)(rbase + r) * N + col] = v > 0.f ? v : 0.f;
        }
    }
}

// ------------------------------------------------------------------
// K2: fold W_link column 1 through the second MLP layer
// ------------------------------------------------------------------
__global__ void prep_link(const float* __restrict__ W_c2, const float* __restrict__ b_c2,
                          const float* __restrict__ W_link, const float* __restrict__ b_link,
                          float* __restrict__ wc2col, float* __restrict__ scal)
{
    int k = blockIdx.x * blockDim.x + threadIdx.x;
    if (k < HID) {
        float s = 0.f;
        for (int d = 0; d < DDIM; ++d) s += W_c2[(size_t)k * DDIM + d] * W_link[2 * d + 1];
        wc2col[k] = s;
    }
    if (blockIdx.x == 0 && threadIdx.x == 0) {
        float c = 0.f;
        for (int d = 0; d < DDIM; ++d) c += b_c2[d] * W_link[2 * d + 1];
        scal[0] = c + b_link[1];
    }
}

// ------------------------------------------------------------------
// K3: col_term[j] = H[j,:]·wc2col + scal[0]   (one wave32 per row)
// ------------------------------------------------------------------
__global__ void colterm_k(const float* __restrict__ H, const float* __restrict__ wc2col,
                          const float* __restrict__ scal, float* __restrict__ colterm)
{
    int wid  = (blockIdx.x * blockDim.x + threadIdx.x) >> 5;
    int lane = threadIdx.x & 31;
    if (wid >= NROW) return;
    const float* hrow = H + (size_t)wid * HID;
    float s = 0.f;
    for (int k = lane; k < HID; k += 32) s += hrow[k] * wc2col[k];
    for (int off = 16; off > 0; off >>= 1) s += __shfl_down(s, off, 32);
    if (lane == 0) colterm[wid] = s + scal[0];
}

// ------------------------------------------------------------------
// K4: row-constant normalization + literal double-threshold
// ------------------------------------------------------------------
__global__ void stats_thresh(const float* __restrict__ colterm, float* __restrict__ t,
                             float* __restrict__ scal)
{
    __shared__ float sd[256];
    __shared__ float mean_s, istd_s;
    const int tid = threadIdx.x;

    float s = 0.f;
    for (int j = tid; j < NROW; j += 256) s += colterm[j];
    sd[tid] = s; __syncthreads();
    for (int o = 128; o > 0; o >>= 1) { if (tid < o) sd[tid] += sd[tid + o]; __syncthreads(); }
    if (tid == 0) mean_s = sd[0] / (float)NROW;
    __syncthreads();
    const float mean = mean_s;

    float ss = 0.f;
    for (int j = tid; j < NROW; j += 256) { float d = colterm[j] - mean; ss += d * d; }
    sd[tid] = ss; __syncthreads();
    for (int o = 128; o > 0; o >>= 1) { if (tid < o) sd[tid] += sd[tid + o]; __syncthreads(); }
    if (tid == 0) istd_s = 1.0f / sqrtf(sd[0] / (float)(NROW - 1));
    __syncthreads();
    const float istd = istd_s;

    float ts = 0.f;
    for (int j = tid; j < NROW; j += 256) {
        float z  = (colterm[j] - mean) * istd;
        float a  = (z > 1.5f) ? z : 0.f;     // where(corr > HI, corr, 0)
        float tv = (a < -1.5f) ? a : 0.f;    // where(adj < LO, adj, 0) -> always 0
        t[j] = tv; ts += tv;
    }
    sd[tid] = ts; __syncthreads();
    for (int o = 128; o > 0; o >>= 1) { if (tid < o) sd[tid] += sd[tid + o]; __syncthreads(); }
    if (tid == 0) scal[1] = sd[0];
}

// ------------------------------------------------------------------
// K5: u[k] = sum_j t[j] * H[j,k]
// ------------------------------------------------------------------
__global__ void th_gemv(const float* __restrict__ t, const float* __restrict__ H,
                        float* __restrict__ u)
{
    int k = blockIdx.x * blockDim.x + threadIdx.x;
    if (k >= HID) return;
    float s = 0.f;
    for (int j = 0; j < NROW; ++j) s += t[j] * H[(size_t)j * HID + k];
    u[k] = s;
}

// ------------------------------------------------------------------
// K6a: v[d] = u·W_c2[:,d] + sum(t)*b_c2[d]
// ------------------------------------------------------------------
__global__ void v_gemv(const float* __restrict__ u, const float* __restrict__ W_c2,
                       const float* __restrict__ b_c2, const float* __restrict__ scal,
                       float* __restrict__ v)
{
    int d = blockIdx.x * blockDim.x + threadIdx.x;
    if (d >= DDIM) return;
    float s = scal[1] * b_c2[d];
    for (int k = 0; k < HID; ++k) s += u[k] * W_c2[(size_t)k * DDIM + d];
    v[d] = s;
}

// ------------------------------------------------------------------
// K6b: outrow[c] = v·W_gcn[:,c] + b_gcn[c]
// ------------------------------------------------------------------
__global__ void outrow_gemv(const float* __restrict__ v, const float* __restrict__ W_gcn,
                            const float* __restrict__ b_gcn, float* __restrict__ outrow)
{
    int c = blockIdx.x * blockDim.x + threadIdx.x;
    if (c >= OUTD) return;
    float s = b_gcn[c];
    for (int d = 0; d < DDIM; ++d) s += v[d] * W_gcn[(size_t)d * OUTD + c];
    outrow[c] = s;
}

// ------------------------------------------------------------------
// K7: broadcast outrow to all 4096 rows (float4 stores; the only real cost)
// ------------------------------------------------------------------
__global__ void broadcast_out(const float* __restrict__ outrow, float4* __restrict__ out)
{
    const int total = NROW * (OUTD / 4);           // 524288 float4
    int idx = blockIdx.x * blockDim.x + threadIdx.x;
    if (idx >= total) return;
    out[idx] = ((const float4*)outrow)[idx & (OUTD / 4 - 1)];
}

// ------------------------------------------------------------------
extern "C" void kernel_launch(void* const* d_in, const int* in_sizes, int n_in,
                              void* d_out, int out_size, void* d_ws, size_t ws_size,
                              hipStream_t stream)
{
    const float* X      = (const float*)d_in[0];   // [1,4096,1024]
    const float* W_c1   = (const float*)d_in[1];   // [1024,512]
    const float* b_c1   = (const float*)d_in[2];   // [512]
    const float* W_c2   = (const float*)d_in[3];   // [512,512]
    const float* b_c2   = (const float*)d_in[4];   // [512]
    const float* W_link = (const float*)d_in[5];   // [1024,2]
    const float* b_link = (const float*)d_in[6];   // [2]
    const float* W_gcn  = (const float*)d_in[7];   // [512,512]
    const float* b_gcn  = (const float*)d_in[8];   // [512]
    float* out = (float*)d_out;

    // ---- workspace carve-up ----
    char* ws = (char*)d_ws;
    size_t off = 0;
    _Float16* Xh  = (_Float16*)(ws + off); off += (size_t)NROW * INDIM * 2;   // 8 MB
    _Float16* Bp  = (_Float16*)(ws + off); off += (size_t)INDIM * HID * 2;    // 1 MB (packed W1)
    float* H      = (float*)(ws + off);    off += (size_t)NROW * HID * 4;     // 8 MB
    float* wc2col = (float*)(ws + off);    off += HID * 4;
    float* colterm= (float*)(ws + off);    off += NROW * 4;
    float* t      = (float*)(ws + off);    off += NROW * 4;
    float* u      = (float*)(ws + off);    off += HID * 4;
    float* v      = (float*)(ws + off);    off += DDIM * 4;
    float* outrow = (float*)(ws + off);    off += OUTD * 4;
    float* scal   = (float*)(ws + off);    off += 256;

    // K0: convert X, convert+pack W_c1
    cvt_f32_f16<<<2048, 256, 0, stream>>>(X, Xh, NROW * INDIM);
    pack_W1<<<(INDIM * HID) / 256, 256, 0, stream>>>(W_c1, Bp);

    // K2 (independent)
    prep_link<<<2, 256, 0, stream>>>(W_c2, b_c2, W_link, b_link, wc2col, scal);

    // K1: WMMA GEMM + bias + relu -> H
    // (4096/16)*(512/64) = 2048 wave32 strips, 8 waves per 256-thread block
    gemm_relu_wmma<<<256, 256, 0, stream>>>(Xh, Bp, b_c1, H);

    // K3..K6: rank-1 tail
    colterm_k<<<NROW / 8, 256, 0, stream>>>(H, wc2col, scal, colterm);
    stats_thresh<<<1, 256, 0, stream>>>(colterm, t, scal);
    th_gemv<<<2, 256, 0, stream>>>(t, H, u);
    v_gemv<<<2, 256, 0, stream>>>(u, W_c2, b_c2, scal, v);
    outrow_gemv<<<2, 256, 0, stream>>>(v, W_gcn, b_gcn, outrow);

    // K7: broadcast to [1,4096,512]
    broadcast_out<<<(NROW * OUTD / 4 + 255) / 256, 256, 0, stream>>>(outrow, (float4*)out);
}
